// DynamicConv_59785944760962
// MI455X (gfx1250) — matrix operand order repaired and checked by reading, not compiled, not run on hardware
//
#include <hip/hip_runtime.h>
#include <hip/hip_fp16.h>

typedef _Float16 v8h  __attribute__((ext_vector_type(8)));
typedef _Float16 v16h __attribute__((ext_vector_type(16)));
typedef float    v8f  __attribute__((ext_vector_type(8)));
typedef float    v4f  __attribute__((ext_vector_type(4)));

#define TAU_F   30.0f
#define KB      4
#define CIN     128
#define COUT    256
#define HID     256
#define BATCH   8
#define HWD     48
#define RED     (CIN * 9)      // 1152 reduction length
#define AROW    1160           // padded A-tile row stride in halves (2320B, 16B aligned, bank-spread)
#define TSTRIDE 68             // per-m row stride (dwords) in transpose scratch ([m][c][k] f32, 16B aligned)
#define TWAVE   1088           // dwords of transpose scratch per wave
#define ATILE_BYTES (48 * AROW * 2)
#define SMEM_BYTES  (ATILE_BYTES + 8 * TWAVE * 4)

// ws layout (bytes)
#define WS_WF16   0            // 1024*1152 f16   = 2,359,296
#define WS_XF16   2359296      // 8*128*48*48 f16 = 4,718,592
#define WS_POOLED 7077888      // 1024 f32
#define WS_ATTN   7081984      // 32 f32

// ---------------------------------------------------------------- prep: pool
__global__ void dyn_pool_kernel(const float* __restrict__ x, float* __restrict__ pooled) {
    __shared__ float red[256];
    const int bc = blockIdx.x;                       // b*128 + c, 0..1023
    const float* p = x + (size_t)bc * (HWD * HWD);
    float s = 0.f;
    for (int i = threadIdx.x; i < HWD * HWD; i += 256) s += p[i];
    red[threadIdx.x] = s;
    __syncthreads();
    for (int off = 128; off > 0; off >>= 1) {
        if (threadIdx.x < off) red[threadIdx.x] += red[threadIdx.x + off];
        __syncthreads();
    }
    if (threadIdx.x == 0) pooled[bc] = red[0] * (1.0f / (HWD * HWD));
}

// ---------------------------------------------------- prep: MLP + 2x softmax
__global__ void dyn_attn_kernel(const float* __restrict__ pooled,
                                const float* __restrict__ w1, const float* __restrict__ b1,
                                const float* __restrict__ w2, const float* __restrict__ b2,
                                float* __restrict__ attn) {
    __shared__ float sp[BATCH * CIN];
    __shared__ float sh[BATCH * HID];
    __shared__ float sl[BATCH * KB];
    for (int i = threadIdx.x; i < BATCH * CIN; i += 256) sp[i] = pooled[i];
    __syncthreads();
    for (int o = threadIdx.x; o < BATCH * HID; o += 256) {
        const int b = o / HID, hh = o % HID;
        float acc = b1[hh];
        for (int c = 0; c < CIN; ++c) acc += sp[b * CIN + c] * w1[hh * CIN + c];
        sh[o] = fmaxf(acc, 0.f);
    }
    __syncthreads();
    if (threadIdx.x < BATCH * KB) {
        const int b = threadIdx.x / KB, k = threadIdx.x % KB;
        float acc = b2[k];
        for (int hh = 0; hh < HID; ++hh) acc += sh[b * HID + hh] * w2[k * HID + hh];
        sl[threadIdx.x] = acc;
    }
    __syncthreads();
    if (threadIdx.x < BATCH) {
        const int b = threadIdx.x;
        float l[KB];
        float m1 = -1e30f;
        for (int k = 0; k < KB; ++k) { l[k] = sl[b * KB + k]; m1 = fmaxf(m1, l[k]); }
        float s1 = 0.f;
        for (int k = 0; k < KB; ++k) { l[k] = expf(l[k] - m1); s1 += l[k]; }
        float m2 = -1e30f;
        for (int k = 0; k < KB; ++k) { l[k] = l[k] / (s1 * TAU_F); m2 = fmaxf(m2, l[k]); }
        float s2 = 0.f;
        float e2[KB];
        for (int k = 0; k < KB; ++k) { e2[k] = expf(l[k] - m2); s2 += e2[k]; }
        for (int k = 0; k < KB; ++k) attn[b * KB + k] = e2[k] / s2;
    }
}

// ----------------------------------------------- prep: generic fp32 -> f16
__global__ void dyn_cvt_kernel(const float* __restrict__ src, _Float16* __restrict__ dst, int total) {
    int idx = blockIdx.x * 1024 + threadIdx.x;
    #pragma unroll
    for (int e = 0; e < 4; ++e, idx += 256)
        if (idx < total) dst[idx] = (_Float16)src[idx];
}

// -------------------------------------------------------------- main kernel
// grid: (48 rows h, 8 images, 2 channel halves); 256 threads = 8 waves.
// Each wave: c-tile of 16 channels; 3 spatial subtiles x 4 branches accumulators.
__global__ __launch_bounds__(256)
void dyn_conv_wmma_kernel(const _Float16* __restrict__ xf16,
                          const _Float16* __restrict__ wf16,
                          const float* __restrict__ convb,
                          const float* __restrict__ attn,
                          float* __restrict__ out) {
    extern __shared__ char smem[];
    _Float16* At = (_Float16*)smem;                      // im2col tile [48 m][1152 r]
    float*    Tws = (float*)(smem + ATILE_BYTES);        // per-wave transpose scratch

    const int h    = blockIdx.x;
    const int img  = blockIdx.y;
    const int cz   = blockIdx.z;
    const int tid  = threadIdx.x;
    const int lane = tid & 31;
    const int wave = tid >> 5;

    // ---- build im2col A tile (f16) in LDS; r = ci*9 + kh*3 + kw (== weight layout)
    // One (ci, m) pair per step; 3x3 window fully unrolled with compile-time kh/kw
    // so only a single div/mod-by-48 is paid per 9 elements.
    const _Float16* xb = xf16 + (size_t)img * CIN * HWD * HWD;
    for (int p = tid; p < CIN * 48; p += 256) {          // 24 iterations, no tail
        const int m  = p % 48;                            // consecutive lanes -> consecutive w
        const int ci = p / 48;
        const _Float16* xc = xb + ci * (HWD * HWD);
        _Float16* arow = At + m * AROW + ci * 9;
        #pragma unroll
        for (int kh = 0; kh < 3; ++kh) {
            const int hh = h + kh - 1;
            const bool vr = (unsigned)hh < HWD;
            const _Float16* rp = xc + hh * HWD;
            #pragma unroll
            for (int kw = 0; kw < 3; ++kw) {
                const int ww = m + kw - 1;
                _Float16 v = (_Float16)0.f;
                if (vr && (unsigned)ww < HWD) v = rp[ww];
                arow[kh * 3 + kw] = v;
            }
        }
    }
    __syncthreads();

    // ---- fragment addressing per ISA layouts (wave32)
    const int laneM = lane & 15;
    const int aoff  = (lane < 16) ? 0 : 8;     // A: halves K 0-7/16-23 vs 8-15/24-31
    const int boff  = (lane < 16) ? 0 : 8;     // B: halves K 0-7/16-23 vs 8-15/24-31
    const int cbase = cz * 128 + wave * 16;

    v8f acc[3][KB] = {};

    for (int ks = 0; ks < RED / 32; ++ks) {
        const int r0 = ks * 32;
        v16h af[3];
        #pragma unroll
        for (int mt = 0; mt < 3; ++mt) {
            const _Float16* ap = At + (mt * 16 + laneM) * AROW + r0 + aoff;
            v8h lo = *(const v8h*)ap;
            v8h hi = *(const v8h*)(ap + 16);
            af[mt] = __builtin_shufflevector(lo, hi, 0,1,2,3,4,5,6,7,8,9,10,11,12,13,14,15);
        }
        v16h bf[KB];
        #pragma unroll
        for (int k = 0; k < KB; ++k) {
            const _Float16* bp = wf16 + (size_t)(k * COUT + cbase + laneM) * RED + r0 + boff;
            v8h lo = *(const v8h*)bp;
            v8h hi = *(const v8h*)(bp + 16);
            bf[k] = __builtin_shufflevector(lo, hi, 0,1,2,3,4,5,6,7,8,9,10,11,12,13,14,15);
            // prefetch next K-step's fragment (L2-resident weights -> WGP$);
            // lands in adjacent ws region at the tail, prefetch is non-faulting
            __builtin_prefetch((const void*)(bp + 32), 0, 3);
        }
        #pragma unroll
        for (int mt = 0; mt < 3; ++mt)
            #pragma unroll
            for (int k = 0; k < KB; ++k)
                acc[mt][k] = __builtin_amdgcn_wmma_f32_16x16x32_f16(
                    false, af[mt], false, bf[k], (short)0, acc[mt][k], false, false);
    }

    // ---- epilogue: bias, per-wave LDS transpose, attn combine, coalesced stores
    float aw[BATCH][KB];
    #pragma unroll
    for (int j = 0; j < BATCH; ++j)
        #pragma unroll
        for (int k = 0; k < KB; ++k) aw[j][k] = attn[j * KB + k];   // uniform -> scalar loads

    const int c_lane = cbase + laneM;
    float bs[KB];
    #pragma unroll
    for (int k = 0; k < KB; ++k) bs[k] = convb[k * COUT + c_lane];
    #pragma unroll
    for (int mt = 0; mt < 3; ++mt)
        #pragma unroll
        for (int k = 0; k < KB; ++k)
            #pragma unroll
            for (int v = 0; v < 8; ++v) acc[mt][k][v] += bs[k];

    float* T = Tws + wave * TWAVE;
    const int mrow = lane >> 4;
    const int mm   = lane & 15;
    #pragma unroll 1
    for (int mt = 0; mt < 3; ++mt) {
        // C layout element (M = v + 8*(lane/16), N = laneM) -> LDS [m][c][k]
        #pragma unroll
        for (int k = 0; k < KB; ++k)
            #pragma unroll
            for (int v = 0; v < 8; ++v)
                T[(v + 8 * mrow) * TSTRIDE + laneM * 4 + k] = acc[mt][k][v];
        // wave-local LDS is in-order: safe to read back without a barrier
        const int w_out = mt * 16 + mm;
        #pragma unroll
        for (int ci = 0; ci < 8; ++ci) {
            const int cc = ci * 2 + mrow;
            v4f t = *(const v4f*)&T[mm * TSTRIDE + cc * 4];
            const int c = cbase + cc;
            #pragma unroll
            for (int j = 0; j < BATCH; ++j) {
                const float val = t.x * aw[j][0] + t.y * aw[j][1] + t.z * aw[j][2] + t.w * aw[j][3];
                out[(((size_t)(img * BATCH + j) * COUT + c) * HWD + h) * HWD + w_out] = val;
            }
        }
    }
}

// ------------------------------------------------------------------- launch
extern "C" void kernel_launch(void* const* d_in, const int* in_sizes, int n_in,
                              void* d_out, int out_size, void* d_ws, size_t ws_size,
                              hipStream_t stream) {
    const float* x     = (const float*)d_in[0];
    const float* convw = (const float*)d_in[1];
    const float* convb = (const float*)d_in[2];
    const float* w1    = (const float*)d_in[3];
    const float* b1    = (const float*)d_in[4];
    const float* w2    = (const float*)d_in[5];
    const float* b2    = (const float*)d_in[6];
    float* out = (float*)d_out;

    char* ws = (char*)d_ws;
    _Float16* wf16   = (_Float16*)(ws + WS_WF16);
    _Float16* xf16   = (_Float16*)(ws + WS_XF16);
    float*    pooled = (float*)(ws + WS_POOLED);
    float*    attn   = (float*)(ws + WS_ATTN);

    dyn_pool_kernel<<<BATCH * CIN, 256, 0, stream>>>(x, pooled);
    dyn_attn_kernel<<<1, 256, 0, stream>>>(pooled, w1, b1, w2, b2, attn);

    const int wtotal = KB * COUT * RED;                  // 1,179,648
    dyn_cvt_kernel<<<(wtotal + 1023) / 1024, 256, 0, stream>>>(convw, wf16, wtotal);
    const int xtotal = BATCH * CIN * HWD * HWD;          // 2,359,296
    dyn_cvt_kernel<<<(xtotal + 1023) / 1024, 256, 0, stream>>>(x, xf16, xtotal);

    hipFuncSetAttribute((const void*)dyn_conv_wmma_kernel,
                        hipFuncAttributeMaxDynamicSharedMemorySize, (int)SMEM_BYTES);
    dim3 grid(HWD, BATCH, 2);
    dyn_conv_wmma_kernel<<<grid, 256, SMEM_BYTES, stream>>>(xf16, wf16, convb, attn, out);
}
